// BPDRLayer_38405597561382
// MI455X (gfx1250) — matrix operand order repaired and checked by reading, not compiled, not run on hardware
//
#include <hip/hip_runtime.h>
#include <hip/hip_bf16.h>
#include <math.h>

#define EMB   128
#define D_IN  192
#define NBAND 3
#define NN    50000
#define NE    320000
#define LAM   0.01f
#define LN_EPS 1e-5f

typedef _Float16 h8v  __attribute__((ext_vector_type(8)));
typedef _Float16 h16v __attribute__((ext_vector_type(16)));
typedef float    f8v  __attribute__((ext_vector_type(8)));

// ---------------- WMMA helper -----------------------------------------------
__device__ __forceinline__ f8v wmma_f16(h16v a, h16v b, f8v c) {
    return __builtin_amdgcn_wmma_f32_16x16x32_f16(false, a, false, b,
                                                  (short)0, c, false, false);
}

// Build 16x32 A-fragment from LDS (f16, rowStride halves). Contiguous k order (not the ISA table order):
// lane: row = lane&15, kb = (lane>>4)*16 ; halves 0..15 -> K=k0+kb+j
__device__ __forceinline__ h16v load_a_frag(const _Float16* base, int rowStride,
                                            int lane, int k0) {
    int row = lane & 15;
    int kb  = (lane >> 4) * 16;
    const h8v* p0 = (const h8v*)(base + row * rowStride + k0 + kb);
    const h8v* p1 = (const h8v*)(base + row * rowStride + k0 + 8 + kb);
    h8v x0 = *p0, x1 = *p1;
    h16v a;
#pragma unroll
    for (int j = 0; j < 8; ++j) { a[j] = x0[j]; a[j + 8] = x1[j]; }
    return a;
}

// Packed B-fragments: frag = 32 lanes * 16 halves contiguous (1024 B)
__device__ __forceinline__ h16v load_b_frag(const _Float16* pack, int fragIdx, int lane) {
    return *(const h16v*)(pack + (size_t)fragIdx * 512 + lane * 16);
}

// ---------------- init: accum = boundary, esum = 0 ---------------------------
__global__ void k_init(const float* __restrict__ bc, float* __restrict__ accum,
                       float* __restrict__ esum) {
    long i = (long)blockIdx.x * blockDim.x + threadIdx.x;
    if (i < (long)NN * EMB) accum[i] = bc[i];
    if (i == 0) *esum = 0.f;
}

// ---------------- fold DFT(192) into band weights, pack to WMMA B layout ----
// A_k = C@r1 - S@i1 (S = -sin), B_k = S@r1 + C@i1, C[t][n]=cos(2pi t n/192)
__global__ void k_pack_stage1(const float* __restrict__ r1, const float* __restrict__ i1,
                              _Float16* __restrict__ apack /* A then B */) {
    int tid = blockIdx.x * blockDim.x + threadIdx.x;   // 2*3*48*512 = 147456
    if (tid >= 2 * NBAND * 48 * 512) return;
    int j    = tid & 15;
    int lane = (tid >> 4) & 31;
    int fg   = tid >> 9;
    int frag = fg % 48;
    int band = (fg / 48) % NBAND;
    int mat  = fg / (48 * NBAND);          // 0 = A, 1 = B
    int ktile = frag >> 3, nt = frag & 7;
    int t = ktile * 32 + ((lane >> 4) * 16) + j;   // B-operand K index
    int m = nt * 16 + (lane & 15);                 // output col
    const float w = 6.283185307179586f / (float)D_IN;
    float acc = 0.f;
    const float* rr = r1 + (size_t)band * D_IN * EMB;
    const float* ii = i1 + (size_t)band * D_IN * EMB;
    for (int n = 0; n < D_IN; ++n) {
        int ph = (t * n) % D_IN;                   // exact phase reduction
        float c = __cosf(w * (float)ph);
        float s = __sinf(w * (float)ph);
        float rv = rr[n * EMB + m], iv = ii[n * EMB + m];
        // A: cos*r1 + sin*i1 ; B: -sin*r1 + cos*i1
        acc += (mat == 0) ? (c * rv + s * iv) : (c * iv - s * rv);
    }
    apack[(size_t)tid] = (_Float16)acc;  // tid already equals linear packed index
}

// ---------------- ifft(384)+fre_w fold: Wc,Ws (f32 scratch) ------------------
__global__ void k_dft_stage2(const float* __restrict__ fre_w,
                             float* __restrict__ Wc, float* __restrict__ Ws) {
    int tid = blockIdx.x * blockDim.x + threadIdx.x;   // 2*384*128 = 98304
    if (tid >= 2 * 384 * EMB) return;
    int m = tid % EMB;
    int n = (tid / EMB) % 384;
    int sel = tid / (384 * EMB);
    const float w = 6.283185307179586f / 384.f;
    float acc = 0.f;
    for (int t = 0; t < 384; ++t) {
        int ph = (n * t) % 384;
        float f = fre_w[t * EMB + m];
        acc += sel ? (-__sinf(w * (float)ph) * f) : (__cosf(w * (float)ph) * f);
    }
    acc *= (1.f / 384.f);
    (sel ? Ws : Wc)[n * EMB + m] = acc;
}

// ---------------- fold comb_w (even rows) into Wc/Ws, pack -------------------
__global__ void k_pack_stage2(const float* __restrict__ Wc, const float* __restrict__ Ws,
                              const float* __restrict__ comb_w,
                              _Float16* __restrict__ wcp, _Float16* __restrict__ wsp) {
    int tid = blockIdx.x * blockDim.x + threadIdx.x;   // 2*96*512 = 98304
    if (tid >= 2 * 96 * 512) return;
    int j    = tid & 15;
    int lane = (tid >> 4) & 31;
    int fg   = tid >> 9;
    int frag = fg % 96;
    int sel  = fg / 96;
    int kt2 = frag >> 3, nt2 = frag & 7;
    int n   = kt2 * 32 + ((lane >> 4) * 16) + j;       // K index (0..383)
    int col = nt2 * 16 + (lane & 15);
    const float* W = sel ? Ws : Wc;
    float acc = 0.f;
    for (int m = 0; m < EMB; ++m)
        acc += W[n * EMB + m] * comb_w[(2 * m) * EMB + col];
    (sel ? wsp : wcp)[(size_t)frag * 512 + lane * 16 + j] = (_Float16)acc;
}

__global__ void k_bias2(const float* __restrict__ fre_b, const float* __restrict__ comb_w,
                        const float* __restrict__ comb_b, float* __restrict__ bias2) {
    int j = threadIdx.x;
    if (j >= EMB) return;
    float acc = comb_b[j];
    for (int m = 0; m < EMB; ++m) acc += fre_b[m] * comb_w[(2 * m) * EMB + j];
    bias2[j] = acc;
}

__global__ void k_pack_lin(const float* __restrict__ lin_w, _Float16* __restrict__ linp) {
    int tid = blockIdx.x * blockDim.x + threadIdx.x;   // 32*512 = 16384
    if (tid >= 32 * 512) return;
    int j = tid & 15, lane = (tid >> 4) & 31, frag = tid >> 9;
    int kt = frag >> 3, nt = frag & 7;
    int t   = kt * 32 + ((lane >> 4) * 16) + j;
    int col = nt * 16 + (lane & 15);
    linp[tid] = (_Float16)lin_w[t * EMB + col];
}

// ---------------- energy (Parseval): energy[e]=192*sum(x^2), + global sum ----
__global__ void k_energy(const float* __restrict__ hidden, const float* __restrict__ ea,
                         const float* __restrict__ et, const int* __restrict__ eidx,
                         float* __restrict__ energy, float* __restrict__ esum) {
    int wid  = threadIdx.x >> 5;
    int lane = threadIdx.x & 31;
    int e = blockIdx.x * 8 + wid;                    // 8 waves/block, 1 edge/wave
    __shared__ float bs;
    if (threadIdx.x == 0) bs = 0.f;
    __syncthreads();
    float s = 0.f;
    int sn = eidx[e];
#pragma unroll
    for (int j = 0; j < 6; ++j) {
        int c = lane + 32 * j;
        float v = (c < 128) ? hidden[(size_t)sn * EMB + c]
                : (c < 160) ? ea[(size_t)e * 32 + (c - 128)]
                            : et[(size_t)e * 32 + (c - 160)];
        s += v * v;
    }
#pragma unroll
    for (int off = 16; off > 0; off >>= 1) s += __shfl_xor(s, off, 32);
    if (lane == 0) {
        float en = 192.f * s;
        energy[e] = en;
        atomicAdd(&bs, en);
    }
    __syncthreads();
    if (threadIdx.x == 0) atomicAdd(esum, bs);
}

__global__ void k_thresholds(const float* __restrict__ esum, const float* __restrict__ alpha,
                             float* __restrict__ thr) {
    float es = *esum;
    for (int k = 0; k < NBAND; ++k) {
        float a = alpha[k];
        float factor = (2.f * (k + 1) - 1.f) / (2.f * NBAND);
        float Q = a * factor * es;
        float b = es / (a * (2.f * NBAND));
        thr[2 * k]     = Q - b;
        thr[2 * k + 1] = Q + b;
    }
}

// ---------------- fused per-edge-tile pipeline (WMMA) ------------------------
// one wave per 16 edges: gather x -> stage1 GEMMs (3 bands) -> mask/bias/relu/
// softshrink -> stage2 GEMMs (fold ifft+fre+comb) -> atomic scatter to accum
__global__ void __launch_bounds__(32)
k_edges(const float* __restrict__ hidden, const float* __restrict__ ea,
        const float* __restrict__ et, const int* __restrict__ eidx,
        const float* __restrict__ energy, const float* __restrict__ thr,
        const float* __restrict__ rb1, const float* __restrict__ ib1,
        const _Float16* __restrict__ apack, const _Float16* __restrict__ bpack,
        const _Float16* __restrict__ wcp, const _Float16* __restrict__ wsp,
        const float* __restrict__ bias2, float* __restrict__ accum) {
    const int e0   = blockIdx.x * 16;
    const int lane = threadIdx.x;
    __shared__ _Float16 xs[16 * D_IN];      // x tile, f16
    __shared__ _Float16 chunk[2 * 16 * 32]; // o_r / o_i chunk, f16
    __shared__ float mkLds[NBAND][16];      // per-(band,row) mask
    __shared__ int   dstLds[16];

    const int* src = eidx;
    const int* dst = eidx + NE;
    if (lane < 16) {
        float en = energy[e0 + lane];
        dstLds[lane] = dst[e0 + lane];
#pragma unroll
        for (int k = 0; k < NBAND; ++k) {
            float lo = thr[2 * k], hi = thr[2 * k + 1];
            mkLds[k][lane] = (en >= lo && en <= hi) ? 1.f : 0.f;
        }
    }

    for (int r = 0; r < 16; ++r) {
        int e = e0 + r;
        int s = src[e];
#pragma unroll
        for (int j = 0; j < 6; ++j) {
            int c = lane + 32 * j;
            float v = (c < 128) ? hidden[(size_t)s * EMB + c]
                    : (c < 160) ? ea[(size_t)e * 32 + (c - 128)]
                                : et[(size_t)e * 32 + (c - 160)];
            xs[r * D_IN + c] = (_Float16)v;
        }
    }
    __syncthreads();

    // hoist all 6 x-tile A-fragments into registers (invariant over bands/cols)
    h16v afrag[6];
#pragma unroll
    for (int kt = 0; kt < 6; ++kt) afrag[kt] = load_a_frag(xs, D_IN, lane, kt * 32);

    f8v acc2[8];
#pragma unroll
    for (int i = 0; i < 8; ++i) acc2[i] = (f8v)0.f;

    const int col = lane & 15;
    const int hi8 = (lane >> 4) * 8;

    for (int k = 0; k < NBAND; ++k) {
        for (int nt = 0; nt < 4; ++nt) {            // 32-col chunk of band output
            // prefetch next chunk's stage-1 B fragments (lane -> distinct 128B line)
            {
                int kk = k, nn = nt + 1;
                if (nn == 4) { nn = 0; kk = (k + 1 < NBAND) ? k + 1 : k; }
                const char* pa = (const char*)(apack + ((size_t)kk * 48 + nn * 2) * 512);
                const char* pb = (const char*)(bpack + ((size_t)kk * 48 + nn * 2) * 512);
#pragma unroll
                for (int q = 0; q < 3; ++q) {
                    __builtin_prefetch(pa + (q * 32 + lane) * 128, 0, 3);
                    __builtin_prefetch(pb + (q * 32 + lane) * 128, 0, 3);
                }
            }
            f8v pr[2], pi[2];
#pragma unroll
            for (int st = 0; st < 2; ++st) { pr[st] = (f8v)0.f; pi[st] = (f8v)0.f; }
#pragma unroll
            for (int kt = 0; kt < 6; ++kt) {        // K over 192
#pragma unroll
                for (int st = 0; st < 2; ++st) {
                    int f = (k * 48) + kt * 8 + (nt * 2 + st);
                    pr[st] = wmma_f16(afrag[kt], load_b_frag(apack, f, lane), pr[st]);
                    pi[st] = wmma_f16(afrag[kt], load_b_frag(bpack, f, lane), pi[st]);
                }
            }
            // mask (per-edge scalar), bias, relu, softshrink -> chunk (f16)
#pragma unroll
            for (int st = 0; st < 2; ++st) {
                int cg = nt * 32 + st * 16 + col;
                float br = rb1[k * EMB + cg];
                float bi = ib1[k * EMB + cg];
#pragma unroll
                for (int v = 0; v < 8; ++v) {
                    int m = v + hi8;
                    float mk = mkLds[k][m];
                    float orv = pr[st][v] * mk + br;
                    orv = fmaxf(orv, 0.f); orv = (orv > LAM) ? orv - LAM : 0.f;
                    float oiv = pi[st][v] * mk + bi;
                    oiv = fmaxf(oiv, 0.f); oiv = (oiv > LAM) ? oiv - LAM : 0.f;
                    chunk[       m * 32 + st * 16 + col] = (_Float16)orv;
                    chunk[512 +  m * 32 + st * 16 + col] = (_Float16)oiv;
                }
            }
            __syncthreads();
            // stage2: msg2 += o_r_chunk @ Wc2[k32] + o_i_chunk @ Ws2[k32]
            h16v ar = load_a_frag(chunk,        32, lane, 0);
            h16v ai = load_a_frag(chunk + 512,  32, lane, 0);
            int kt2 = k * 4 + nt;
#pragma unroll
            for (int nt2 = 0; nt2 < 8; ++nt2) {
                acc2[nt2] = wmma_f16(ar, load_b_frag(wcp, kt2 * 8 + nt2, lane), acc2[nt2]);
                acc2[nt2] = wmma_f16(ai, load_b_frag(wsp, kt2 * 8 + nt2, lane), acc2[nt2]);
            }
            __syncthreads();
        }
    }
    // scatter: accum[dst] += msg2 + bias2
#pragma unroll
    for (int nt2 = 0; nt2 < 8; ++nt2) {
        int cg = nt2 * 16 + col;
        float b2 = bias2[cg];
#pragma unroll
        for (int v = 0; v < 8; ++v) {
            int m = v + hi8;
            atomicAdd(&accum[(size_t)dstLds[m] * EMB + cg], acc2[nt2][v] + b2);
        }
    }
}

// ---------------- node stage: lin GEMM + LayerNorm + ReLU --------------------
__global__ void __launch_bounds__(32)
k_nodes(const float* __restrict__ accum, const _Float16* __restrict__ linp,
        const float* __restrict__ lin_b, const float* __restrict__ ln_g,
        const float* __restrict__ ln_b, float* __restrict__ out) {
    const int n0   = blockIdx.x * 16;
    const int lane = threadIdx.x;
    __shared__ _Float16 xsh[16 * EMB];
    __shared__ float    ysh[16 * EMB];
    for (int r = 0; r < 16; ++r) {
#pragma unroll
        for (int j = 0; j < 4; ++j) {
            int c = lane + 32 * j;
            xsh[r * EMB + c] = (_Float16)accum[(size_t)(n0 + r) * EMB + c];
        }
    }
    __syncthreads();
    h16v afrag[4];
#pragma unroll
    for (int kt = 0; kt < 4; ++kt) afrag[kt] = load_a_frag(xsh, EMB, lane, kt * 32);
    f8v acc[8];
#pragma unroll
    for (int i = 0; i < 8; ++i) acc[i] = (f8v)0.f;
#pragma unroll
    for (int kt = 0; kt < 4; ++kt) {
#pragma unroll
        for (int nt = 0; nt < 8; ++nt)
            acc[nt] = wmma_f16(afrag[kt], load_b_frag(linp, kt * 8 + nt, lane), acc[nt]);
    }
    const int col = lane & 15;
    const int hi8 = (lane >> 4) * 8;
#pragma unroll
    for (int nt = 0; nt < 8; ++nt) {
        int cg = nt * 16 + col;
        float bb = lin_b[cg];
#pragma unroll
        for (int v = 0; v < 8; ++v) ysh[(v + hi8) * EMB + cg] = acc[nt][v] + bb;
    }
    __syncthreads();
    if (lane < 16) {
        int row = lane;
        float mu = 0.f;
        for (int c = 0; c < EMB; ++c) mu += ysh[row * EMB + c];
        mu *= (1.f / EMB);
        float var = 0.f;
        for (int c = 0; c < EMB; ++c) {
            float d = ysh[row * EMB + c] - mu;
            var += d * d;
        }
        var *= (1.f / EMB);
        float rstd = rsqrtf(var + LN_EPS);
        for (int c = 0; c < EMB; ++c) {
            float v = (ysh[row * EMB + c] - mu) * rstd * ln_g[c] + ln_b[c];
            out[(size_t)(n0 + row) * EMB + c] = fmaxf(v, 0.f);
        }
    }
}

// ---------------- launcher ---------------------------------------------------
extern "C" void kernel_launch(void* const* d_in, const int* in_sizes, int n_in,
                              void* d_out, int out_size, void* d_ws, size_t ws_size,
                              hipStream_t stream) {
    const float* hidden  = (const float*)d_in[0];
    const float* ea      = (const float*)d_in[1];
    const float* et      = (const float*)d_in[2];
    const float* bc      = (const float*)d_in[3];
    const float* alpha   = (const float*)d_in[4];
    const float* r1      = (const float*)d_in[5];
    const float* i1      = (const float*)d_in[6];
    const float* rb1     = (const float*)d_in[7];
    const float* ib1     = (const float*)d_in[8];
    const float* fre_w   = (const float*)d_in[9];
    const float* fre_b   = (const float*)d_in[10];
    const float* comb_w  = (const float*)d_in[11];
    const float* comb_b  = (const float*)d_in[12];
    const float* lin_w   = (const float*)d_in[13];
    const float* lin_b   = (const float*)d_in[14];
    const float* ln_g    = (const float*)d_in[15];
    const float* ln_b    = (const float*)d_in[16];
    const int*   eidx    = (const int*)d_in[17];
    float* out = (float*)d_out;

    char* ws = (char*)d_ws;
    size_t o = 0;
    float* accum  = (float*)(ws + o); o += (size_t)NN * EMB * 4;
    float* energy = (float*)(ws + o); o += (size_t)NE * 4;
    float* esum   = (float*)(ws + o); o += 256;
    float* thr    = (float*)(ws + o); o += 256;
    float* Wc     = (float*)(ws + o); o += 384 * EMB * 4;
    float* Ws     = (float*)(ws + o); o += 384 * EMB * 4;
    _Float16* apack = (_Float16*)(ws + o); o += (size_t)NBAND * 48 * 512 * 2; // A
    _Float16* bpack = (_Float16*)(ws + o); o += (size_t)NBAND * 48 * 512 * 2; // B
    _Float16* wcp   = (_Float16*)(ws + o); o += 96 * 512 * 2;
    _Float16* wsp   = (_Float16*)(ws + o); o += 96 * 512 * 2;
    float* bias2    = (float*)(ws + o); o += 512;
    _Float16* linp  = (_Float16*)(ws + o); o += 32 * 512 * 2;
    (void)ws_size; (void)n_in; (void)in_sizes; (void)out_size;

    // init accum = boundary, esum = 0
    k_init<<<(NN * EMB + 255) / 256, 256, 0, stream>>>(bc, accum, esum);
    // weight folding / packing (cheap, deterministic each call)
    k_pack_stage1<<<(2 * NBAND * 48 * 512 + 255) / 256, 256, 0, stream>>>(r1, i1, apack);
    k_dft_stage2<<<(2 * 384 * EMB + 255) / 256, 256, 0, stream>>>(fre_w, Wc, Ws);
    k_pack_stage2<<<(2 * 96 * 512 + 255) / 256, 256, 0, stream>>>(Wc, Ws, comb_w, wcp, wsp);
    k_bias2<<<1, 128, 0, stream>>>(fre_b, comb_w, comb_b, bias2);
    k_pack_lin<<<(32 * 512 + 255) / 256, 256, 0, stream>>>(lin_w, linp);
    // energy (Parseval) + global reduce, then band thresholds
    k_energy<<<NE / 8, 256, 0, stream>>>(hidden, ea, et, eidx, energy, esum);
    k_thresholds<<<1, 1, 0, stream>>>(esum, alpha, thr);
    // fused edge pipeline (the WMMA heavyweight)
    k_edges<<<NE / 16, 32, 0, stream>>>(hidden, ea, et, eidx, energy, thr,
                                        rb1, ib1, apack, bpack, wcp, wsp, bias2, accum);
    // node GEMM + LayerNorm + ReLU
    k_nodes<<<NN / 16, 32, 0, stream>>>(accum, linp, lin_b, ln_g, ln_b, out);
}
